// UltraDeepGAT_83193516524088
// MI455X (gfx1250) — matrix operand order repaired and checked by reading, not compiled, not run on hardware
//
#include <hip/hip_runtime.h>
#include <hip/hip_bf16.h>
#include <math.h>

// ---------------------------------------------------------------------------
// UltraDeepGAT forward for MI455X (gfx1250, wave32, WMMA).
// Dense GEMMs: LDS-staged A tile (f16), v_wmma_f32_16x16x32_f16, two 16x16
// output tiles per wave. Edge phase: wave-per-edge coalesced gathers + f32
// atomics (lane = channel, C == wave32).
// ---------------------------------------------------------------------------

typedef __attribute__((ext_vector_type(16))) _Float16 v16h;
typedef __attribute__((ext_vector_type(8)))  float    v8f;

#define DDIM 256
#define HEADS 8
#define CH 32
#define EPSV 1e-5f

// ---- order-preserving float <-> uint encoding for atomic segment-max ------
__device__ __forceinline__ unsigned ordF(float f) {
    unsigned u = __float_as_uint(f);
    return (u & 0x80000000u) ? ~u : (u | 0x80000000u);
}
__device__ __forceinline__ float unordF(unsigned u) {
    unsigned v = (u & 0x80000000u) ? (u ^ 0x80000000u) : ~u;
    return __uint_as_float(v);
}

__device__ __forceinline__ float lrelu(float v, float s) {
    return v > 0.0f ? v : s * v;
}

__device__ __forceinline__ float waveRedAdd(float v) {
    #pragma unroll
    for (int m = 16; m > 0; m >>= 1) v += __shfl_xor(v, m, 32);
    return v;
}

// ---------------------------------------------------------------------------
// Generic zero-fill (also used for the uint max buffer: 0 < ordF(-inf)).
// ---------------------------------------------------------------------------
__global__ void zero_kernel(float* __restrict__ p, size_t n) {
    size_t i = (size_t)blockIdx.x * blockDim.x + threadIdx.x;
    if (i < n) p[i] = 0.0f;
}

// ---------------------------------------------------------------------------
// WMMA GEMM: Out[M x Nc] = epilogue(A[M x K] @ W[K x Nc] + bias)
// MODE 0: +bias                     (xl / xr transforms)
// MODE 1: +bias, BN(eval), lrelu0.1 (input proj, h1)
// MODE 2: +bias, lrelu0.1           (h2)
//
// Block = 256 threads = 8 waves; block computes a 16-row x 256-col strip.
// A row-block (16 x K f32) is converted to f16 once into LDS (shared by all
// 8 waves -> 8x less A traffic); each wave owns a 16x32 column strip with
// two f32 accumulators sharing one A fragment.
//
// Fragment layouts per CDNA5 ISA 7.12.2:
//   A (16x32 f16): lane = kh*16+m ; elems 0..7 -> K = k0+kh*8+i,
//                                   elems 8..15 -> K = k0+16+kh*8+i
//   B (32x16 f16): lane = kh*16+n ; elems -> K as in A; column colBase+n (+16 for bf1)
//   D (16x16 f32): lane = kh*16+n ; VGPR v -> row rowBase + kh*8 + v
// LDS fill + barrier happens before the wave-uniform column guard, so every
// thread reaches __syncthreads() and EXEC is all-1s at every WMMA.
// ---------------------------------------------------------------------------
template <int MODE>
__global__ void __launch_bounds__(256) wmma_gemm_kernel(
        const float* __restrict__ A, const float* __restrict__ W,
        const float* __restrict__ bias,
        const float* __restrict__ bn_g, const float* __restrict__ bn_b,
        float* __restrict__ Out, int K, int Nc) {
    __shared__ _Float16 As[16][264];   // stride 264 halves -> bank-conflict-free
    const int tid = threadIdx.x;
    const int rowBase = blockIdx.x * 16;

    // Cooperative A tile load: 16 rows x K, f32 -> f16.
    const int chunks = K >> 4;                       // 16-half chunks per row
    if (tid < 16 * chunks) {
        const int row = tid / chunks;
        const int ch  = tid - row * chunks;
        const float* src = A + (size_t)(rowBase + row) * K + ch * 16;
        _Float16* dst = &As[row][ch * 16];
        #pragma unroll
        for (int i = 0; i < 16; ++i) dst[i] = (_Float16)src[i];
    }
    __syncthreads();

    const int wave = tid >> 5;
    const int lane = tid & 31;
    const int colBase = wave * 32;
    if (colBase < Nc) {                              // wave-uniform
        const int m16 = lane & 15;
        const int kh  = lane >> 4;
        const _Float16* __restrict__ ar = &As[m16][0];

        v8f acc0 = {}, acc1 = {};
        for (int k0 = 0; k0 < K; k0 += 32) {
            v16h af, bf0, bf1;
            #pragma unroll
            for (int i = 0; i < 8; ++i) {            // two ds_load_b128 runs
                af[i]     = ar[k0 + kh * 8 + i];
                af[8 + i] = ar[k0 + 16 + kh * 8 + i];
            }
            const float* pb = W + (size_t)k0 * Nc + colBase + m16;
            #pragma unroll
            for (int i = 0; i < 8; ++i) {
                bf0[i]     = (_Float16)pb[(size_t)(kh * 8 + i) * Nc];
                bf0[8 + i] = (_Float16)pb[(size_t)(16 + kh * 8 + i) * Nc];
                bf1[i]     = (_Float16)pb[(size_t)(kh * 8 + i) * Nc + 16];
                bf1[8 + i] = (_Float16)pb[(size_t)(16 + kh * 8 + i) * Nc + 16];
            }
            acc0 = __builtin_amdgcn_wmma_f32_16x16x32_f16(
                false, af, false, bf0, (short)0, acc0, false, false);
            acc1 = __builtin_amdgcn_wmma_f32_16x16x32_f16(
                false, af, false, bf1, (short)0, acc1, false, false);
        }

        const float bnscale = rsqrtf(1.0f + EPSV);
        #pragma unroll
        for (int t = 0; t < 2; ++t) {
            const v8f acc = t ? acc1 : acc0;
            const int n = colBase + t * 16 + m16;
            const float bs = bias[n];
            const float gg = (MODE == 1) ? bn_g[n] : 0.0f;
            const float bb = (MODE == 1) ? bn_b[n] : 0.0f;
            #pragma unroll
            for (int v = 0; v < 8; ++v) {
                float val = acc[v] + bs;
                if (MODE == 1) val = val * bnscale * gg + bb;
                if (MODE >= 1) val = lrelu(val, 0.1f);
                Out[(size_t)(rowBase + kh * 8 + v) * Nc + n] = val;
            }
        }
    }
}

// ---------------------------------------------------------------------------
// LayerNorm: one wave per 256-wide row, 8 rows / block.
// ---------------------------------------------------------------------------
__global__ void __launch_bounds__(256) layernorm_kernel(
        const float* __restrict__ x, const float* __restrict__ g,
        const float* __restrict__ b, float* __restrict__ y, int N) {
    const int lane = threadIdx.x & 31;
    const int row  = blockIdx.x * 8 + (threadIdx.x >> 5);
    if (row >= N) return;
    const float* p = x + (size_t)row * DDIM;
    float v[8]; float s = 0.0f;
    #pragma unroll
    for (int t = 0; t < 8; ++t) { v[t] = p[t * 32 + lane]; s += v[t]; }
    s = waveRedAdd(s);
    const float mu = s * (1.0f / DDIM);
    float q = 0.0f;
    #pragma unroll
    for (int t = 0; t < 8; ++t) { float d = v[t] - mu; q += d * d; }
    q = waveRedAdd(q);
    const float inv = rsqrtf(q * (1.0f / DDIM) + EPSV);
    float* o = y + (size_t)row * DDIM;
    #pragma unroll
    for (int t = 0; t < 8; ++t) {
        int f = t * 32 + lane;
        o[f] = (v[t] - mu) * inv * g[f] + b[f];
    }
}

// ---------------------------------------------------------------------------
// Edge pass 1: wave per edge. lane = channel c, iteration = head h.
// logit[e][h] = sum_c att[h][c] * lrelu(xl[src]+xr[dst], 0.2); atomic seg-max.
// ---------------------------------------------------------------------------
__global__ void __launch_bounds__(256) edge_logits_kernel(
        const float* __restrict__ xl, const float* __restrict__ xr,
        const float* __restrict__ att, const int* __restrict__ ei,
        int E, int N, float* __restrict__ logit, unsigned* __restrict__ mmax) {
    const int wid  = (int)(((size_t)blockIdx.x * blockDim.x + threadIdx.x) >> 5);
    const int lane = threadIdx.x & 31;
    const int Etot = E + N;
    if (wid >= Etot) return;
    int s, d;
    if (wid < E) { s = ei[wid]; d = ei[E + wid]; } else { s = d = wid - E; }
    const float* ps = xl + (size_t)s * DDIM;
    const float* pd = xr + (size_t)d * DDIM;
    #pragma unroll
    for (int h = 0; h < HEADS; ++h) {
        float v = ps[h * CH + lane] + pd[h * CH + lane];
        v = lrelu(v, 0.2f);
        float t = waveRedAdd(v * att[h * CH + lane]);
        if (lane == 0) {
            logit[(size_t)wid * HEADS + h] = t;
            atomicMax(&mmax[(size_t)d * HEADS + h], ordF(t));
        }
    }
}

// ---------------------------------------------------------------------------
// Edge pass 2: exp(logit - max), scatter den and weighted xl[src] into agg.
// ---------------------------------------------------------------------------
__global__ void __launch_bounds__(256) edge_agg_kernel(
        const float* __restrict__ xl, const float* __restrict__ logit,
        const unsigned* __restrict__ mmax, const int* __restrict__ ei,
        int E, int N, float* __restrict__ den, float* __restrict__ agg) {
    const int wid  = (int)(((size_t)blockIdx.x * blockDim.x + threadIdx.x) >> 5);
    const int lane = threadIdx.x & 31;
    const int Etot = E + N;
    if (wid >= Etot) return;
    int s, d;
    if (wid < E) { s = ei[wid]; d = ei[E + wid]; } else { s = d = wid - E; }
    const float* ps = xl + (size_t)s * DDIM;
    float exh[HEADS];
    #pragma unroll
    for (int h = 0; h < HEADS; ++h) {
        float lg = logit[(size_t)wid * HEADS + h];
        float mx = unordF(mmax[(size_t)d * HEADS + h]);
        exh[h] = __expf(lg - mx);
    }
    if (lane < HEADS) atomicAdd(&den[(size_t)d * HEADS + lane], exh[lane]);
    #pragma unroll
    for (int h = 0; h < HEADS; ++h) {
        atomicAdd(&agg[(size_t)d * DDIM + h * CH + lane],
                  ps[h * CH + lane] * exh[h]);
    }
}

// ---------------------------------------------------------------------------
// Node update: o = lrelu(agg/den + conv_b, 0.1); h += scale*o;
// hsum += softmax(scale_weights)[layer] * h
// ---------------------------------------------------------------------------
__global__ void __launch_bounds__(256) node_update_kernel(
        const float* __restrict__ agg, const float* __restrict__ den,
        const float* __restrict__ conv_b, const float* __restrict__ scales,
        const float* __restrict__ scale_w, int layer, int L,
        float* __restrict__ h, float* __restrict__ hsum, int N) {
    size_t i = (size_t)blockIdx.x * blockDim.x + threadIdx.x;
    if (i >= (size_t)N * DDIM) return;
    const int n  = (int)(i >> 8);
    const int f  = (int)(i & 255);
    const int hd = f >> 5;
    float o = agg[i] / den[(size_t)n * HEADS + hd] + conv_b[f];
    o = lrelu(o, 0.1f);
    float hv = h[i] + scales[layer] * o;
    h[i] = hv;
    float mx = scale_w[0];
    for (int l = 1; l < L; ++l) mx = fmaxf(mx, scale_w[l]);
    float se = 0.0f;
    for (int l = 0; l < L; ++l) se += __expf(scale_w[l] - mx);
    hsum[i] += (__expf(scale_w[layer] - mx) / se) * hv;
}

// ---------------------------------------------------------------------------
// Head: out[row] = dot(y2[row, 0:128], h3_w) + h3_b. Wave per row.
// ---------------------------------------------------------------------------
__global__ void __launch_bounds__(256) head_kernel(
        const float* __restrict__ y2, const float* __restrict__ w,
        const float* __restrict__ b, float* __restrict__ out, int N) {
    const int lane = threadIdx.x & 31;
    const int row  = blockIdx.x * 8 + (threadIdx.x >> 5);
    if (row >= N) return;
    const float* p = y2 + (size_t)row * 128;
    float s = p[lane]      * w[lane]
            + p[lane + 32] * w[lane + 32]
            + p[lane + 64] * w[lane + 64]
            + p[lane + 96] * w[lane + 96];
    s = waveRedAdd(s);
    if (lane == 0) out[row] = s + b[0];
}

// ---------------------------------------------------------------------------
// Host-side orchestration
// ---------------------------------------------------------------------------
template <int MODE>
static void launch_gemm(const float* A, const float* W, const float* bias,
                        const float* g, const float* b, float* Out,
                        int M, int K, int Nc, hipStream_t s) {
    hipLaunchKernelGGL((wmma_gemm_kernel<MODE>), dim3(M / 16), dim3(256), 0, s,
                       A, W, bias, g, b, Out, K, Nc);
}

static void launch_zero(float* p, size_t n, hipStream_t s) {
    hipLaunchKernelGGL(zero_kernel, dim3((unsigned)((n + 255) / 256)),
                       dim3(256), 0, s, p, n);
}

extern "C" void kernel_launch(void* const* d_in, const int* in_sizes, int n_in,
                              void* d_out, int out_size, void* d_ws, size_t ws_size,
                              hipStream_t stream) {
    const float* x       = (const float*)d_in[0];
    const int*   ei      = (const int*)d_in[1];
    const float* win_w   = (const float*)d_in[2];
    const float* win_b   = (const float*)d_in[3];
    const float* bn1_g   = (const float*)d_in[4];
    const float* bn1_b   = (const float*)d_in[5];
    const float* ln_g    = (const float*)d_in[6];
    const float* ln_b    = (const float*)d_in[7];
    const float* Wl      = (const float*)d_in[8];
    const float* bl      = (const float*)d_in[9];
    const float* Wr      = (const float*)d_in[10];
    const float* br      = (const float*)d_in[11];
    const float* att     = (const float*)d_in[12];
    const float* conv_b  = (const float*)d_in[13];
    const float* scales  = (const float*)d_in[14];
    const float* scale_w = (const float*)d_in[15];
    const float* h1_w    = (const float*)d_in[16];
    const float* h1_b    = (const float*)d_in[17];
    const float* bn2_g   = (const float*)d_in[18];
    const float* bn2_b   = (const float*)d_in[19];
    const float* h2_w    = (const float*)d_in[20];
    const float* h2_b    = (const float*)d_in[21];
    const float* h3_w    = (const float*)d_in[22];
    const float* h3_b    = (const float*)d_in[23];

    const int N = in_sizes[0] / 128;       // 50000
    const int E = in_sizes[1] / 2;         // 800000
    const int L = in_sizes[14];            // 6
    const int IN = 128;
    const int Etot = E + N;

    // Workspace layout (floats). hn is reused as agg and as y1; xr as y2.
    float* ws = (float*)d_ws;
    const size_t nd = (size_t)N * DDIM;
    float*    h     = ws;
    float*    hsum  = ws + nd;
    float*    hn    = ws + 2 * nd;                 // also agg, also y1
    float*    xl    = ws + 3 * nd;
    float*    xr    = ws + 4 * nd;                 // also y2 (N x 128)
    float*    logit = ws + 5 * nd;                 // Etot * 8
    unsigned* mmax  = (unsigned*)(ws + 5 * nd + (size_t)Etot * HEADS);
    float*    den   = ws + 5 * nd + (size_t)Etot * HEADS + (size_t)N * HEADS;

    const int edgeBlocks = (Etot + 7) / 8;         // 8 waves (edges) per block
    const int rowBlocks  = (N + 7) / 8;
    const int ndBlocks   = (int)((nd + 255) / 256);

    // h = lrelu(BN(x @ win_w + win_b), 0.1)
    launch_gemm<1>(x, win_w, win_b, bn1_g, bn1_b, h, N, IN, DDIM, stream);
    launch_zero(hsum, nd, stream);

    for (int i = 0; i < L; ++i) {
        hipLaunchKernelGGL(layernorm_kernel, dim3(rowBlocks), dim3(256), 0, stream,
                           h, ln_g + i * DDIM, ln_b + i * DDIM, hn, N);
        launch_gemm<0>(hn, Wl + (size_t)i * DDIM * DDIM, bl + i * DDIM,
                       nullptr, nullptr, xl, N, DDIM, DDIM, stream);
        launch_gemm<0>(hn, Wr + (size_t)i * DDIM * DDIM, br + i * DDIM,
                       nullptr, nullptr, xr, N, DDIM, DDIM, stream);
        // zero mmax+den (contiguous) and agg (reuses hn, free after GEMMs)
        launch_zero((float*)mmax, (size_t)N * HEADS * 2, stream);
        launch_zero(hn, nd, stream);
        hipLaunchKernelGGL(edge_logits_kernel, dim3(edgeBlocks), dim3(256), 0, stream,
                           xl, xr, att + (size_t)i * HEADS * CH, ei, E, N,
                           logit, mmax);
        hipLaunchKernelGGL(edge_agg_kernel, dim3(edgeBlocks), dim3(256), 0, stream,
                           xl, logit, mmax, ei, E, N, den, /*agg=*/hn);
        hipLaunchKernelGGL(node_update_kernel, dim3(ndBlocks), dim3(256), 0, stream,
                           /*agg=*/hn, den, conv_b + i * DDIM, scales, scale_w,
                           i, L, h, hsum, N);
    }

    // y1 = lrelu(BN(hsum @ h1_w + h1_b), 0.1)   (y1 aliases hn)
    launch_gemm<1>(hsum, h1_w, h1_b, bn2_g, bn2_b, hn, N, DDIM, DDIM, stream);
    // y2 = lrelu(y1 @ h2_w + h2_b, 0.1)          (y2 aliases xr)
    launch_gemm<2>(hn, h2_w, h2_b, nullptr, nullptr, xr, N, DDIM, DDIM / 2, stream);
    // out = y2 @ h3_w + h3_b
    hipLaunchKernelGGL(head_kernel, dim3(rowBlocks), dim3(256), 0, stream,
                       xr, h3_w, h3_b, (float*)d_out, N);
}
